// LineVectorizer_31275951850357
// MI455X (gfx1250) — matrix operand order repaired and checked by reading, not compiled, not run on hardware
//
#include <hip/hip_runtime.h>
#include <hip/hip_bf16.h>

// ---------------------------------------------------------------------------
// Types for CDNA5 WMMA (wave32): bf16 A/B fragments, f32 accumulator.
// ---------------------------------------------------------------------------
typedef __bf16  v16bf __attribute__((ext_vector_type(16)));
typedef __bf16  v8bf  __attribute__((ext_vector_type(8)));
typedef __bf16  v2bf  __attribute__((ext_vector_type(2)));
typedef float   v8f   __attribute__((ext_vector_type(8)));

#define HW      256
#define HW2     65536        // 256*256
#define CIN     256
#define DLOI    128
#define B_SZ    4
#define NLINES  5000
#define NROWS   (B_SZ * NLINES)   // 20000
#define DFC     1024
#define KFEAT   1024              // DLOI * N_PTS1

#define LDS_STRIDE 40             // bf16 elems: 80B rows -> 16B aligned, no bank conflicts

// ---------------------------------------------------------------------------
// A-fragment loader: row-major bf16 matrix, 16x32 tile (contiguous k order:
// lane<16 holds k = k0+{0..15}; lane>=16 holds k0+16+{0..15}).
// Two contiguous 16B loads per lane.
// ---------------------------------------------------------------------------
__device__ inline v16bf load_a_frag(const __bf16* __restrict__ A, int row,
                                    int ldk, int k0, int lane) {
    const int base = k0 + ((lane >> 4) << 4);
    const __bf16* p = A + (size_t)row * ldk + base;
    v8bf lo = *(const v8bf*)(p);
    v8bf hi = *(const v8bf*)(p + 8);
    v16bf f;
#pragma unroll
    for (int i = 0; i < 8; ++i) { f[i] = lo[i]; f[i + 8] = hi[i]; }
    return f;
}

__device__ inline v16bf combine16(v8bf lo, v8bf hi) {
    v16bf f;
#pragma unroll
    for (int i = 0; i < 8; ++i) { f[i] = lo[i]; f[i + 8] = hi[i]; }
    return f;
}

__device__ inline v8f wmma_bf16(v16bf a, v16bf b, v8f c) {
    return __builtin_amdgcn_wmma_f32_16x16x32_bf16(
        /*neg_a=*/false, a, /*neg_b=*/false, b,
        /*c_mod=*/(short)0, c, /*reuse_a=*/false, /*reuse_b=*/false);
}

// ---------------------------------------------------------------------------
// Prep: f32 -> bf16 convert (fc1_w) and f32 [K][N] -> bf16 [N][K] transpose.
// ---------------------------------------------------------------------------
__global__ __launch_bounds__(256) void k_cvt_bf16(const float* __restrict__ s,
                                                  __bf16* __restrict__ d, int n) {
    int i = blockIdx.x * 256 + threadIdx.x;
    if (i < n) d[i] = (__bf16)s[i];
}

__global__ __launch_bounds__(256) void k_transpose_bf16(const float* __restrict__ w,
                                                        __bf16* __restrict__ wt,
                                                        int K, int N) {
    int i = blockIdx.x * 256 + threadIdx.x;
    if (i < K * N) {
        int k = i / N, n = i - k * N;
        wt[(size_t)n * K + k] = (__bf16)w[i];
    }
}

// ---------------------------------------------------------------------------
// fc1: x[b,d,p] = sum_c fc1_w[d,c] * F[b,c,p] + fc1_b[d]   (stored as bf16)
// Block = 256 thr (8 waves) -> M=128 (wave w owns rows 16w..16w+15) x N=64 px.
// The 32c x 64px feature tile is staged TRANSPOSED in LDS ([p][k], stride 40)
// so each 32x16 B-fragment is two ds_load_b128 per lane (conflict-free).
// Feature reads are non-temporal: streamed once, keep L2 for the bf16 map.
// ---------------------------------------------------------------------------
__global__ __launch_bounds__(256) void k_fc1_gemm(const float* __restrict__ F,
                                                  const __bf16* __restrict__ W,
                                                  const float* __restrict__ bias,
                                                  __bf16* __restrict__ X) {
    __shared__ __bf16 ldsB[64 * LDS_STRIDE];
    const int tid  = threadIdx.x;
    const int lane = tid & 31, wave = tid >> 5;
    const int b    = blockIdx.y;
    const int p0   = blockIdx.x * 64;
    const int m0   = wave * 16;
    const size_t fbase = (size_t)b * CIN * HW2;
    const int khalf = lane >> 4;

    v8f acc[4] = {};
    for (int ks = 0; ks < CIN / 32; ++ks) {
        const int c0 = ks * 32;
        __syncthreads();
        // cooperative transposed fill: thread handles (pixel p, k-pair kp),
        // packs 2 bf16 into one 32-bit LDS store.  Global reads coalesced.
#pragma unroll
        for (int i = 0; i < 4; ++i) {
            int idx = tid + i * 256;               // 0..1023
            int p = idx & 63, kp = idx >> 6;       // p:0..63  kp:0..15
            float f0 = __builtin_nontemporal_load(
                F + fbase + (size_t)(c0 + 2 * kp) * HW2 + p0 + p);
            float f1 = __builtin_nontemporal_load(
                F + fbase + (size_t)(c0 + 2 * kp + 1) * HW2 + p0 + p);
            v2bf pk; pk[0] = (__bf16)f0; pk[1] = (__bf16)f1;
            *(v2bf*)(&ldsB[p * LDS_STRIDE + 2 * kp]) = pk;
        }
        __syncthreads();

        v16bf a = load_a_frag(W, m0 + (lane & 15), CIN, c0, lane);
#pragma unroll
        for (int j = 0; j < 4; ++j) {
            const __bf16* bp =
                &ldsB[((lane & 15) + j * 16) * LDS_STRIDE + khalf * 16];
            v8bf lo = *(const v8bf*)(bp);          // ds_load_b128
            v8bf hi = *(const v8bf*)(bp + 8);      // ds_load_b128
            acc[j] = wmma_bf16(a, combine16(lo, hi), acc[j]);
        }
    }
    // D layout: lane col = lane&15; VGPR v -> row v + 8*(lane>>4)
    const int col = lane & 15, rofs = khalf * 8;
#pragma unroll
    for (int v = 0; v < 8; ++v) {
        const int row = m0 + rofs + v;
        const float bv = bias[row];
#pragma unroll
        for (int j = 0; j < 4; ++j) {
            X[((size_t)b * DLOI + row) * HW2 + p0 + j * 16 + col] =
                (__bf16)(acc[j][v] + bv);
        }
    }
}

// ---------------------------------------------------------------------------
// Line sampling + maxpool.  One wave per line: lane p (0..31) owns sample
// point p; bilinear-gather over 128 channels from the L2-resident bf16 map,
// then quad-max (__shfl_xor) pools 32 -> 8 points.  feats flattened D-major,
// P-minor to match torch's reshape.
// ---------------------------------------------------------------------------
__global__ __launch_bounds__(256) void k_sample(const __bf16* __restrict__ X,
                                                const float* __restrict__ lines,
                                                __bf16* __restrict__ feats) {
    const int lane = threadIdx.x & 31, wave = threadIdx.x >> 5;
    const int L = blockIdx.x * 8 + wave;            // 0..19999 (grid exact)
    const int b = L / NLINES;
    const float* ln = lines + (size_t)L * 4;        // [e0y,e0x,e1y,e1x]
    const float lam = (float)lane * (1.0f / 31.0f);
    const float px = ln[0] * lam + ln[2] * (1.0f - lam) - 0.5f;
    const float py = ln[1] * lam + ln[3] * (1.0f - lam) - 0.5f;
    const float px0 = fminf(fmaxf(floorf(px), 0.0f), 255.0f);
    const float py0 = fminf(fmaxf(floorf(py), 0.0f), 255.0f);
    const float px1 = fminf(px0 + 1.0f, 255.0f);
    const float py1 = fminf(py0 + 1.0f, 255.0f);
    const int ix0 = (int)px0, iy0 = (int)py0, ix1 = (int)px1, iy1 = (int)py1;
    const float w00 = (px1 - px) * (py1 - py);
    const float w10 = (px - px0) * (py1 - py);
    const float w01 = (px1 - px) * (py - py0);
    const float w11 = (px - px0) * (py - py0);
    const int o00 = ix0 * HW + iy0, o10 = ix1 * HW + iy0;
    const int o01 = ix0 * HW + iy1, o11 = ix1 * HW + iy1;
    const size_t xbase = (size_t)b * DLOI * HW2;
    __bf16* frow = feats + (size_t)L * KFEAT;
    const int  p1 = lane >> 2;
    const bool writer = (lane & 3) == 0;

#pragma unroll 4
    for (int d = 0; d < DLOI; ++d) {
        const __bf16* xd = X + xbase + (size_t)d * HW2;
        float v = (float)xd[o00] * w00 + (float)xd[o10] * w10 +
                  (float)xd[o01] * w01 + (float)xd[o11] * w11;
        v = fmaxf(v, __shfl_xor(v, 1));
        v = fmaxf(v, __shfl_xor(v, 2));
        if (writer) frow[d * 8 + p1] = (__bf16)v;       // D-major, P-minor
    }
}

// ---------------------------------------------------------------------------
// MLP layer: H = relu(A @ W + b).  A: [M][K] bf16 row-major, Wt: [N][K] bf16
// (pre-transposed so each lane's 32x16 B-fragment is one contiguous 32B load).
// Block = 8 waves; each wave owns TWO M-tiles x 4 N-tiles (8 accumulators),
// so every B-fragment feeds two WMMAs: 24B of loads per WMMA in steady state.
// Requires M % 32 == 0 (M = 20000 -> 625 exact tile-pairs).
// ---------------------------------------------------------------------------
__global__ __launch_bounds__(256) void k_mlp_gemm(const __bf16* __restrict__ A,
                                                  const __bf16* __restrict__ Wt,
                                                  const float* __restrict__ bias,
                                                  __bf16* __restrict__ H,
                                                  int M, int K, int N) {
    const int tid = threadIdx.x, lane = tid & 31, wave = tid >> 5;
    const int pair = blockIdx.x * 8 + wave;
    if (pair * 32 >= M) return;                     // wave-uniform exit
    const int m0 = pair * 32;
    const int n0 = blockIdx.y * 64;
    const int row0 = m0 + (lane & 15);
    const int khalf = lane >> 4;

    v8f acc[2][4] = {};
    for (int k0 = 0; k0 < K; k0 += 32) {
        v16bf a0 = load_a_frag(A, row0,      K, k0, lane);
        v16bf a1 = load_a_frag(A, row0 + 16, K, k0, lane);
#pragma unroll
        for (int j = 0; j < 4; ++j) {
            const __bf16* bp =
                Wt + (size_t)(n0 + j * 16 + (lane & 15)) * K + k0 + khalf * 16;
            v16bf bf = *(const v16bf*)bp;           // contiguous 16 k-values
            acc[0][j] = wmma_bf16(a0, bf, acc[0][j]);
            acc[1][j] = wmma_bf16(a1, bf, acc[1][j]);
        }
    }
    const int col = lane & 15, rofs = khalf * 8;
#pragma unroll
    for (int j = 0; j < 4; ++j) {
        const float bv = bias[n0 + j * 16 + col];
#pragma unroll
        for (int t = 0; t < 2; ++t) {
#pragma unroll
            for (int v = 0; v < 8; ++v) {
                const float r = acc[t][j][v] + bv;
                H[(size_t)(m0 + t * 16 + rofs + v) * N + n0 + j * 16 + col] =
                    (__bf16)fmaxf(r, 0.0f);
            }
        }
    }
}

// ---------------------------------------------------------------------------
// Final 1024 -> 3 layer: one wave per output row, shuffle reduction.
// ---------------------------------------------------------------------------
__global__ __launch_bounds__(256) void k_final(const __bf16* __restrict__ H,
                                               const float* __restrict__ W3,
                                               const float* __restrict__ b3,
                                               float* __restrict__ out, int M) {
    const int lane = threadIdx.x & 31, wave = threadIdx.x >> 5;
    const int r = blockIdx.x * 8 + wave;
    if (r >= M) return;
    const __bf16* hr = H + (size_t)r * DFC;
    float a0 = 0.f, a1 = 0.f, a2 = 0.f;
    for (int k = lane; k < DFC; k += 32) {
        const float h = (float)hr[k];
        const float* w = W3 + k * 3;
        a0 += h * w[0]; a1 += h * w[1]; a2 += h * w[2];
    }
#pragma unroll
    for (int s = 16; s > 0; s >>= 1) {
        a0 += __shfl_down(a0, s);
        a1 += __shfl_down(a1, s);
        a2 += __shfl_down(a2, s);
    }
    if (lane == 0) {
        out[r * 3 + 0] = a0 + b3[0];
        out[r * 3 + 1] = a1 + b3[1];
        out[r * 3 + 2] = a2 + b3[2];
    }
}

// ---------------------------------------------------------------------------
// Host launcher.  Inputs (setup_inputs order):
//  0 feature [4,256,256,256] f32   1 lines [4,5000,2,2] f32
//  2 fc1_w [128,256]   3 fc1_b [128]
//  4 w1 [1024,1024]    5 b1 [1024]   6 w2 [1024,1024]   7 b2 [1024]
//  8 w3 [1024,3]       9 b3 [3]
// ---------------------------------------------------------------------------
extern "C" void kernel_launch(void* const* d_in, const int* in_sizes, int n_in,
                              void* d_out, int out_size, void* d_ws, size_t ws_size,
                              hipStream_t stream) {
    const float* feature = (const float*)d_in[0];
    const float* lines   = (const float*)d_in[1];
    const float* fc1_w   = (const float*)d_in[2];
    const float* fc1_b   = (const float*)d_in[3];
    const float* w1      = (const float*)d_in[4];
    const float* b1      = (const float*)d_in[5];
    const float* w2      = (const float*)d_in[6];
    const float* b2      = (const float*)d_in[7];
    const float* w3      = (const float*)d_in[8];
    const float* b3      = (const float*)d_in[9];
    float* out = (float*)d_out;

    // Workspace layout (bytes); h1 aliases the fc1 map (dead after sampling).
    char* ws = (char*)d_ws;
    const size_t x_bytes    = (size_t)B_SZ * DLOI * HW2 * 2;   // 67,108,864
    const size_t feat_bytes = (size_t)NROWS * KFEAT * 2;       // 40,960,000
    __bf16* x_bf    = (__bf16*)(ws);
    __bf16* feats   = (__bf16*)(ws + x_bytes);
    __bf16* h1      = (__bf16*)(ws);                           // alias of x
    __bf16* h2      = (__bf16*)(ws + x_bytes + feat_bytes);
    __bf16* fc1w_bf = (__bf16*)(ws + x_bytes + 2 * feat_bytes);
    __bf16* w1t     = (__bf16*)(ws + x_bytes + 2 * feat_bytes + 65536);
    __bf16* w2t     = (__bf16*)(ws + x_bytes + 2 * feat_bytes + 65536 + 2097152);

    // Prep: bf16 weight conversions / transposes (tiny, one-shot).
    k_cvt_bf16<<<dim3((DLOI * CIN + 255) / 256), dim3(256), 0, stream>>>(
        fc1_w, fc1w_bf, DLOI * CIN);
    k_transpose_bf16<<<dim3((DFC * DFC + 255) / 256), dim3(256), 0, stream>>>(
        w1, w1t, KFEAT, DFC);
    k_transpose_bf16<<<dim3((DFC * DFC + 255) / 256), dim3(256), 0, stream>>>(
        w2, w2t, DFC, DFC);

    // fc1 1x1-conv GEMM -> bf16 feature map (fits in L2).
    k_fc1_gemm<<<dim3(HW2 / 64, B_SZ), dim3(256), 0, stream>>>(
        feature, fc1w_bf, fc1_b, x_bf);

    // Line sampling + maxpool -> feats [20000 x 1024] bf16.
    k_sample<<<dim3(NROWS / 8), dim3(256), 0, stream>>>(x_bf, lines, feats);

    // MLP (WMMA): h1 = relu(feats@w1+b1); h2 = relu(h1@w2+b2).
    const int pairs = NROWS / 32;                   // 625 exact
    dim3 ggrid((pairs + 7) / 8, DFC / 64);          // (79, 16)
    k_mlp_gemm<<<ggrid, dim3(256), 0, stream>>>(feats, w1t, b1, h1,
                                                NROWS, KFEAT, DFC);
    k_mlp_gemm<<<ggrid, dim3(256), 0, stream>>>(h1, w2t, b2, h2,
                                                NROWS, DFC, DFC);

    // logits = h2 @ w3 + b3.
    k_final<<<dim3(NROWS / 8), dim3(256), 0, stream>>>(h2, w3, b3, out, NROWS);
}
